// ConceptEmbedding_47253230190842
// MI455X (gfx1250) — compile-verified, matrix operand order checked by reading, not run on hardware
//
#include <hip/hip_runtime.h>
#include <hip/hip_bf16.h>
#include <stdint.h>

constexpr int MROWS = 8192;   // B*S
constexpr int KDIM  = 8192;   // C
constexpr int NDIM  = 256;    // D
constexpr int KCENT = 512;    // centroids
constexpr float FREEDOM = 0.1f;
constexpr int KSTEPS = KDIM / 32;   // 256

typedef __attribute__((ext_vector_type(16))) __bf16        v16bf;
typedef __attribute__((ext_vector_type(2)))  __bf16        v2bf;
typedef __attribute__((ext_vector_type(8)))  float         v8f;
typedef __attribute__((ext_vector_type(8)))  unsigned int  v8u;

// f32 -> bf16 pair (lo in [15:0], hi in [31:16]).
#if __has_builtin(__builtin_amdgcn_cvt_pk_bf16_f32)
__device__ __forceinline__ unsigned int pack_bf16(float lo, float hi) {
  v2bf r = __builtin_amdgcn_cvt_pk_bf16_f32(lo, hi);
  return __builtin_bit_cast(unsigned, r);
}
#else
// round-to-nearest (+0x8000) then one v_perm_b32 to gather the high halves:
// result bytes {0,1} = lo bytes {2,3}; bytes {2,3} = hi bytes {2,3} (sel 0x07060302)
__device__ __forceinline__ unsigned int pack_bf16(float lo, float hi) {
  unsigned ulo = __builtin_bit_cast(unsigned, lo) + 0x8000u;
  unsigned uhi = __builtin_bit_cast(unsigned, hi) + 0x8000u;
  return __builtin_amdgcn_perm(uhi, ulo, 0x07060302u);
}
#endif

__device__ __forceinline__ v8f wmma_bf16(v8u a, v8u b, v8f c) {
  return __builtin_amdgcn_wmma_f32_16x16x32_bf16(
      false, __builtin_bit_cast(v16bf, a),
      false, __builtin_bit_cast(v16bf, b),
      (short)0, c, false, false);
}

// Per-lane fragment: 8 dwords contiguous so two b128 loads land in place.
union AFrag { uint4 q[2]; v8u v; };
struct BFrag { union { uint4 q[4]; v8u v[2]; } u; };

// kp_local for (hi, j): lanes 0-15 take K-pairs {0..3, 8..11}, lanes 16-31 {4..7, 12..15}
__device__ __forceinline__ int kp_local(int hi, int j) {
  return (j < 4) ? (hi * 4 + j) : (8 + hi * 4 + (j - 4));
}

// ---------------------------------------------------------------------------
// K0a: pack concept_emb [K=8192][N=256] f32 -> bf16 K-pair dwords in per-lane
//      fragment order: embp[step(256)][ntile(16)][lane(32)][j(8)]
// ---------------------------------------------------------------------------
__global__ __launch_bounds__(256)
void k0_pack_emb(const float* __restrict__ emb, unsigned* __restrict__ embp) {
  const int tid = blockIdx.x * 256 + threadIdx.x;     // 1,048,576 total
  const int j  = tid & 7;
  const int l  = (tid >> 3) & 31;
  const int nt = (tid >> 8) & 15;
  const int s  = tid >> 12;
  const int col = nt * 16 + (l & 15);
  const int k   = s * 32 + 2 * kp_local(l >> 4, j);
  embp[tid] = pack_bf16(emb[(size_t)k * NDIM + col],
                        emb[(size_t)(k + 1) * NDIM + col]);
}

// ---------------------------------------------------------------------------
// K0b: pack centroids (as B = cent^T) into fragment order:
//      cpkf[ks(8)][ntile(32)][lane(32)][j(8)]
// ---------------------------------------------------------------------------
__global__ __launch_bounds__(256)
void k0_pack_cent(const float* __restrict__ cent, unsigned* __restrict__ cpkf) {
  const int tid = blockIdx.x * 256 + threadIdx.x;     // 65,536 total
  const int j  = tid & 7;
  const int l  = (tid >> 3) & 31;
  const int nt = (tid >> 8) & 31;
  const int ks = tid >> 13;
  const int col = nt * 16 + (l & 15);                 // centroid id
  const int k   = ks * 32 + 2 * kp_local(l >> 4, j);  // position along D
  cpkf[tid] = pack_bf16(cent[(size_t)col * NDIM + k],
                        cent[(size_t)col * NDIM + k + 1]);
}

// ---------------------------------------------------------------------------
// K0c: per-centroid squared norms.
// ---------------------------------------------------------------------------
__global__ __launch_bounds__(128)
void k0_norms(const float* __restrict__ cent, float* __restrict__ cnorm) {
  __shared__ float red[128];
  const int b = blockIdx.x, t = threadIdx.x;
  float2 v = *(const float2*)(cent + (size_t)b * NDIM + 2 * t);
  red[t] = v.x * v.x + v.y * v.y;
  __syncthreads();
  for (int s = 64; s > 0; s >>= 1) {
    if (t < s) red[t] += red[t + s];
    __syncthreads();
  }
  if (t == 0) cnorm[b] = red[0];
}

// ---------------------------------------------------------------------------
// K1: e_free = (seq / rowsum(seq)) @ emb  via bf16 WMMA, f32 accum.
// 256 thr (8 waves) per block; 64 rows x 256 cols. Wave w owns N-tiles
// {2w,2w+1} x M-tiles 0..3, so each B fragment is loaded exactly once per
// block (register double-buffered from the L2-resident prepacked embp).
// A: f32 streamed from HBM once, converted + row-summed in-register, staged
// in a double-buffered LDS tile (fragment-ready, pitch 20 dw) -> one barrier
// per K-step.
// ---------------------------------------------------------------------------
__global__ __launch_bounds__(256)
void k1_gemm_norm(const float* __restrict__ A, const unsigned* __restrict__ embp,
                  float* __restrict__ out) {
  __shared__ unsigned lA[2][64 * 20];   // [buf][row][kp 0..15], pitch 20 dw (80B)
  __shared__ float lsum[64 * 8];
  __shared__ float linv[64];

  const int t    = threadIdx.x;
  const int lane = t & 31;
  const int wave = t >> 5;
  const int l16  = lane & 15;
  const int hi   = lane >> 4;

  // A staging: 8 threads per row, 4 floats each; two rows per thread
  const int r0   = t >> 3;
  const int r1   = r0 + 32;
  const int koff = (t & 7) * 4;
  const long rowg0 = (long)blockIdx.x * 64 + r0;
  const long rowg1 = rowg0 + 32;

  float4 a0, a1;
  float ps0 = 0.f, ps1 = 0.f;

  auto loadGA = [&](int s) {
    const long kk = (long)s * 32;
    a0 = *(const float4*)(A + rowg0 * KDIM + kk + koff);
    a1 = *(const float4*)(A + rowg1 * KDIM + kk + koff);
    ps0 += a0.x + a0.y + a0.z + a0.w;      // fused row-count accumulation
    ps1 += a1.x + a1.y + a1.z + a1.w;
  };
  auto storeLA = [&](int buf) {
    uint2 pa0 = { pack_bf16(a0.x, a0.y), pack_bf16(a0.z, a0.w) };
    uint2 pa1 = { pack_bf16(a1.x, a1.y), pack_bf16(a1.z, a1.w) };
    *(uint2*)&lA[buf][r0 * 20 + (t & 7) * 2] = pa0;
    *(uint2*)&lA[buf][r1 * 20 + (t & 7) * 2] = pa1;
  };
  auto loadB = [&](int s, BFrag& f) {      // 2 N-tiles, 32B contiguous per lane
    const unsigned* p = embp + ((((size_t)s * 16) + 2 * wave) * 32 + lane) * 8;
    f.u.q[0] = *(const uint4*)(p);
    f.u.q[1] = *(const uint4*)(p + 4);
    f.u.q[2] = *(const uint4*)(p + 256);   // next N-tile (+32 lanes * 8 dw)
    f.u.q[3] = *(const uint4*)(p + 260);
  };

  const v8f vzero = {0.f, 0.f, 0.f, 0.f, 0.f, 0.f, 0.f, 0.f};
  v8f acc[8];
#pragma unroll
  for (int i = 0; i < 8; ++i) acc[i] = vzero;

  auto compute = [&](int buf, const BFrag& f) {
#pragma unroll
    for (int mt = 0; mt < 4; ++mt) {
      const unsigned* pa = &lA[buf][(16 * mt + l16) * 20 + hi * 4];
      AFrag af;
      af.q[0] = *(const uint4*)(pa);       // K-pairs {0..3 | 4..7}
      af.q[1] = *(const uint4*)(pa + 8);   // K-pairs {8..11 | 12..15}
      acc[mt * 2 + 0] = wmma_bf16(af.v, f.u.v[0], acc[mt * 2 + 0]);
      acc[mt * 2 + 1] = wmma_bf16(af.v, f.u.v[1], acc[mt * 2 + 1]);
    }
  };

  BFrag cb, nb;
  loadGA(0);
  storeLA(0);
  loadB(0, cb);
  __syncthreads();

#pragma unroll 1
  for (int s = 0; s < KSTEPS; s += 2) {
    // even step s: WMMAs read buf0+cb while s+1 global loads fly
    loadGA(s + 1);
    loadB(s + 1, nb);
    if (s + 2 < KSTEPS)
      __builtin_prefetch(A + rowg0 * KDIM + (long)(s + 2) * 32 + koff, 0, 1);
    compute(0, cb);
    storeLA(1);                // pack+store s+1 into idle buffer (no conflict)
    __syncthreads();
    // odd step s+1
    const bool more = (s + 2) < KSTEPS;
    if (more) {
      loadGA(s + 2);
      loadB(s + 2, cb);
    }
    compute(1, nb);
    if (more) storeLA(0);
    __syncthreads();
  }

  // Row counts -> 1/cnt (cnt==0 -> 1)
  lsum[r0 * 8 + (t & 7)] = ps0;
  lsum[r1 * 8 + (t & 7)] = ps1;
  __syncthreads();
  if (t < 64) {
    float c = 0.f;
#pragma unroll
    for (int j = 0; j < 8; ++j) c += lsum[t * 8 + j];
    linv[t] = 1.0f / ((c == 0.0f) ? 1.0f : c);
  }
  __syncthreads();

#pragma unroll
  for (int mt = 0; mt < 4; ++mt) {
#pragma unroll
    for (int q = 0; q < 2; ++q) {
      const int col = (2 * wave + q) * 16 + l16;
#pragma unroll
      for (int v = 0; v < 8; ++v) {
        const int rl = 16 * mt + v + 8 * hi;   // C/D layout: VGPR v -> M=v (+8 hi)
        out[((long)blockIdx.x * 64 + rl) * NDIM + col] = acc[mt * 2 + q][v] * linv[rl];
      }
    }
  }
}

// ---------------------------------------------------------------------------
// K2: G = X @ cent^T (bf16 WMMA), argmin_k(||c_k||^2 - 2G), blend in-place.
// Block: 256 thr (8 waves), 16 rows x all 512 centroids (64 per wave).
// X staged in LDS in fragment order [ks][lane][8] (pitch 12 dw).
// ---------------------------------------------------------------------------
__global__ __launch_bounds__(256)
void k2_centroid_blend(const float* __restrict__ cent,
                       const unsigned* __restrict__ cpkf,
                       const float* __restrict__ cnorm,
                       float* __restrict__ out) {
  __shared__ unsigned lXf[8 * 32 * 12];
  __shared__ unsigned long long slot[16];

  const int t = threadIdx.x;
  const int lane = t & 31, wave = t >> 5, l16 = lane & 15, hi = lane >> 4;
  const long m0 = (long)blockIdx.x * 16;
  const int row = t >> 4;
  const int c0  = (t & 15) * 16;

  // Load our 16 f32 of e_free; keep in regs for the final blend.
  float4 x[4];
#pragma unroll
  for (int i = 0; i < 4; ++i)
    x[i] = *(const float4*)(out + (m0 + row) * NDIM + c0 + 4 * i);

  unsigned pr[8] = {
    pack_bf16(x[0].x, x[0].y), pack_bf16(x[0].z, x[0].w),
    pack_bf16(x[1].x, x[1].y), pack_bf16(x[1].z, x[1].w),
    pack_bf16(x[2].x, x[2].y), pack_bf16(x[2].z, x[2].w),
    pack_bf16(x[3].x, x[3].y), pack_bf16(x[3].z, x[3].w) };
#pragma unroll
  for (int j = 0; j < 8; ++j) {           // scatter into fragment-order LDS
    const int kpg = c0 / 2 + j;           // global K-pair 0..127
    const int ks  = kpg >> 4;
    const int kpl = kpg & 15;
    const int hit = (kpl >> 2) & 1;
    const int jj  = (kpl & 3) + ((kpl >> 3) << 2);
    lXf[((ks * 32) + row + 16 * hit) * 12 + jj] = pr[j];
  }
  if (t < 16) slot[t] = ~0ull;
  __syncthreads();

  const v8f vzero = {0.f, 0.f, 0.f, 0.f, 0.f, 0.f, 0.f, 0.f};
  v8f acc[4] = { vzero, vzero, vzero, vzero };
#pragma unroll 1
  for (int ks = 0; ks < 8; ++ks) {
    const unsigned* pX = lXf + ((size_t)ks * 32 + lane) * 12;
    AFrag af;
    af.q[0] = *(const uint4*)(pX);
    af.q[1] = *(const uint4*)(pX + 4);
#pragma unroll
    for (int p = 0; p < 4; ++p) {
      const unsigned* pB = cpkf + (((size_t)ks * 32 + wave * 4 + p) * 32 + lane) * 8;
      BFrag bf;
      bf.u.q[0] = *(const uint4*)(pB);
      bf.u.q[1] = *(const uint4*)(pB + 4);
      acc[p] = wmma_bf16(af.v, bf.u.v[0], acc[p]);
    }
  }

  const int colbase = wave * 64;
  float cn[4];
#pragma unroll
  for (int p = 0; p < 4; ++p) cn[p] = cnorm[colbase + p * 16 + l16];
#pragma unroll
  for (int v = 0; v < 8; ++v) {
    float best = __FLT_MAX__;
    int bcol = 0;
#pragma unroll
    for (int p = 0; p < 4; ++p) {
      const float s = cn[p] - 2.0f * acc[p][v];
      if (s < best) { best = s; bcol = colbase + p * 16 + l16; }
    }
    unsigned su = __builtin_bit_cast(unsigned, best);
    su = (su & 0x80000000u) ? ~su : (su | 0x80000000u);    // float -> sortable u32
    const unsigned long long key = ((unsigned long long)su << 32) | (unsigned)bcol;
    atomicMin(&slot[v + 8 * hi], key);                     // ds_min_u64 argmin
  }
  __syncthreads();

  const unsigned idx = (unsigned)(slot[row] & 0xFFFFFFFFu);
#pragma unroll
  for (int i = 0; i < 4; ++i) {
    float4 c = *(const float4*)(cent + (size_t)idx * NDIM + c0 + 4 * i);
    float4 o;
    o.x = FREEDOM * x[i].x + (1.0f - FREEDOM) * c.x;
    o.y = FREEDOM * x[i].y + (1.0f - FREEDOM) * c.y;
    o.z = FREEDOM * x[i].z + (1.0f - FREEDOM) * c.z;
    o.w = FREEDOM * x[i].w + (1.0f - FREEDOM) * c.w;
    *(float4*)(out + (m0 + row) * NDIM + c0 + 4 * i) = o;
  }
}

// ---------------------------------------------------------------------------
extern "C" void kernel_launch(void* const* d_in, const int* in_sizes, int n_in,
                              void* d_out, int out_size, void* d_ws, size_t ws_size,
                              hipStream_t stream) {
  (void)in_sizes; (void)n_in; (void)out_size; (void)ws_size;
  const float* seq  = (const float*)d_in[0];   // [8192][8192]
  const float* emb  = (const float*)d_in[1];   // [8192][256]
  const float* cent = (const float*)d_in[2];   // [512][256]
  float* out = (float*)d_out;                  // [8192][256]

  // ws: embp 4 MB | cpkf 256 KB | cnorm 2 KB
  unsigned* embp = (unsigned*)d_ws;
  unsigned* cpkf = (unsigned*)((char*)d_ws + (size_t)KSTEPS * 16 * 32 * 8 * 4);
  float*   cnorm = (float*)((char*)cpkf + (size_t)8 * 32 * 32 * 8 * 4);

  k0_pack_emb <<<KSTEPS * 16 * 32 * 8 / 256, 256, 0, stream>>>(emb, embp);
  k0_pack_cent<<<8 * 32 * 32 * 8 / 256,      256, 0, stream>>>(cent, cpkf);
  k0_norms    <<<KCENT, 128, 0, stream>>>(cent, cnorm);
  k1_gemm_norm<<<MROWS / 64, 256, 0, stream>>>(seq, embp, out);
  k2_centroid_blend<<<MROWS / 16, 256, 0, stream>>>(cent, cpkf, cnorm, out);
}
